// LambdaNetRegressor_48833778155616
// MI455X (gfx1250) — matrix operand order repaired and checked by reading, not compiled, not run on hardware
//
#include <hip/hip_runtime.h>
#include <hip/hip_bf16.h>

// ---------------------------------------------------------------------------
// LambdaNet regressor, fused per-batch with CDNA5 WMMA (gfx1250, wave32).
//   B=1024, S=256, D_IN=64, H=256, K=64
// Algebra (removes values/attn materialization entirely):
//   E   = X @ W_emb + b_emb                       (S x H)
//   Sc  = E @ W_k + b_k ; keys = softmax_S(Sc)    (S x K)
//   Q   = E @ W_q + b_q                           (S x K)
//   G   = keys^T @ E                              (K x H)
//   CTX = G @ W_v + b_v        (since sum_s keys[s,k] == 1)
//   M   = Q^T @ Wd_mat         (Wd reshaped S x H)
//   out = <CTX, M> + b_d
// ---------------------------------------------------------------------------

typedef _Float16 half_t;
typedef __attribute__((ext_vector_type(16))) _Float16 v16h;
typedef __attribute__((ext_vector_type(8)))  _Float16 v8h;
typedef __attribute__((ext_vector_type(4)))  _Float16 v4h;
typedef __attribute__((ext_vector_type(8)))  float    v8f;

#define BB   1024
#define SS   256
#define DIN  64
#define HH   256
#define KK   64

// f16 transposed-weight workspace layout (element offsets)
#define WS_WEMBT 0        // (H=256 x DIN=64)
#define WS_WKT   16384    // (K=64  x H=256)
#define WS_WQT   32768    // (K=64  x H=256)
#define WS_WVT   49152    // (H=256 x H=256)
#define WS_WDT   114688   // (H=256 x S=256)
#define WS_TOTAL 180224

// ---------------------------------------------------------------------------
// Weight prep: fp32 -> f16, transposed to (N x Kc) so B-fragments are
// contiguous along the contraction dim.
// ---------------------------------------------------------------------------
__global__ void lambda_prep_weights(const float* __restrict__ We,
                                    const float* __restrict__ Wk,
                                    const float* __restrict__ Wq,
                                    const float* __restrict__ Wv,
                                    const float* __restrict__ Wd,
                                    half_t* __restrict__ ws) {
  int i = blockIdx.x * blockDim.x + threadIdx.x;
  if (i >= WS_TOTAL) return;
  if (i < WS_WKT) {                       // WembT[n][k] = W_emb[k][n]
    int n = i >> 6, k = i & 63;
    ws[i] = (half_t)We[k * HH + n];
  } else if (i < WS_WQT) {                // WkT[n][h] = W_k[h][n]
    int j = i - WS_WKT; int n = j >> 8, h = j & 255;
    ws[i] = (half_t)Wk[h * KK + n];
  } else if (i < WS_WVT) {                // WqT[n][h] = W_q[h][n]
    int j = i - WS_WQT; int n = j >> 8, h = j & 255;
    ws[i] = (half_t)Wq[h * KK + n];
  } else if (i < WS_WDT) {                // WvT[n][h] = W_v[h][n]
    int j = i - WS_WVT; int n = j >> 8, h = j & 255;
    ws[i] = (half_t)Wv[h * HH + n];
  } else {                                // WdT[h][s] = W_d[s*H + h]
    int j = i - WS_WDT; int h = j >> 8, s = j & 255;
    ws[i] = (half_t)Wd[s * HH + h];
  }
}

// ---------------------------------------------------------------------------
// Fragment loaders per CDNA5 ISA 7.12.2 VGPR layouts (wave32).
// A (16x32 f16, M x Kc): lane m=lane&15, hi=lane>>4.
//   elems 0..7  = A[m][k0 + hi*8 + 0..7]
//   elems 8..15 = A[m][k0 + 16 + hi*8 + 0..7]
// ---------------------------------------------------------------------------
__device__ __forceinline__ v16h frag_a(const half_t* base, int ld,
                                       int row0, int k0, int lane) {
  int m = lane & 15, hi = (lane >> 4) & 1;
  const half_t* p = base + (row0 + m) * ld + k0 + hi * 8;
  v8h lo = *(const v8h*)(p);
  v8h up = *(const v8h*)(p + 16);
  return __builtin_shufflevector(lo, up, 0,1,2,3,4,5,6,7,8,9,10,11,12,13,14,15);
}

// B (32x16 f16, Kc x N) from TRANSPOSED storage baseT[n][k] (N x Kc row-major):
//   lane n=lane&15, hi=lane>>4 ; elems 0..7 = B[k0 + hi*8 + 0..7][n], elems 8..15 = B[k0 + 16 + hi*8 + 0..7][n]
__device__ __forceinline__ v16h frag_bT(const half_t* baseT, int ld,
                                        int n0, int k0, int lane) {
  int n = lane & 15, hi = (lane >> 4) & 1;
  const half_t* p = baseT + (n0 + n) * ld + k0 + hi * 8;
  v8h lo = *(const v8h*)(p);
  v8h up = *(const v8h*)(p + 16);
  return __builtin_shufflevector(lo, up, 0,1,2,3,4,5,6,7,8,9,10,11,12,13,14,15);
}

// D/C (16x16 f32): lane n=lane&15, hi=lane>>4; reg r -> row r + hi*8, col n.
__device__ __forceinline__ void store_d_f16(half_t* out, int ld, int row0,
                                            int col0, const v8f& d, int lane) {
  int n = lane & 15, hi = (lane >> 4) & 1;
#pragma unroll
  for (int r = 0; r < 8; ++r)
    out[(row0 + r + hi * 8) * ld + col0 + n] = (half_t)d[r];
}
__device__ __forceinline__ void store_d_f16_T(half_t* out, int ld, int row0,
                                              int col0, const v8f& d, int lane) {
  int n = lane & 15, hi = (lane >> 4) & 1;
#pragma unroll
  for (int r = 0; r < 8; ++r)
    out[(col0 + n) * ld + (row0 + r + hi * 8)] = (half_t)d[r];
}
__device__ __forceinline__ void store_d_f32(float* out, int ld, int row0,
                                            int col0, const v8f& d, int lane) {
  int n = lane & 15, hi = (lane >> 4) & 1;
#pragma unroll
  for (int r = 0; r < 8; ++r)
    out[(row0 + r + hi * 8) * ld + col0 + n] = d[r];
}

#define WMMA_F16(acc, a, b) \
  acc = __builtin_amdgcn_wmma_f32_16x16x32_f16(false, (a), false, (b), (short)0, (acc), false, false)

// ---------------------------------------------------------------------------
// Main fused kernel: one workgroup (8 wave32) per batch element.
// Dynamic LDS layout (bytes), phases reuse dead regions:
//   [0,      131072) E    f16 256x256           (P1..P4c)
//   [131072, 163840) Xh   f16 256x64  (P0-P1) | CTX f16 64x256 (P5b..P6)
//   [131072, 196608) Sc   f32 256x64  (P2-P3) | ETb f16 128x256 (P4b..P5a)
//   [196608, 229376) KT   f16 64x256  (keys^T)
//   [229376, 262144) QT   f16 64x256  (Q^T)
//   [262144, 294912) G    f16 64x256
//   [294912, 294976) red  f32 x8
// ---------------------------------------------------------------------------
__global__ __launch_bounds__(256)
void lambda_fused_kernel(const float* __restrict__ X,      // (B,S,DIN)
                         const float* __restrict__ b_emb,  // (H)
                         const float* __restrict__ b_k,    // (K)
                         const float* __restrict__ b_q,    // (K)
                         const float* __restrict__ b_v,    // (H)
                         const float* __restrict__ b_d,    // (1)
                         const half_t* __restrict__ ws,    // f16 weights
                         float* __restrict__ out) {        // (B)
  extern __shared__ char smem[];
  half_t* E   = (half_t*)(smem);
  half_t* Xh  = (half_t*)(smem + 131072);
  float*  Sc  = (float*)(smem + 131072);
  half_t* ETb = (half_t*)(smem + 131072);   // 128x256 transpose bounce
  half_t* CTX = (half_t*)(smem + 131072);
  half_t* KT  = (half_t*)(smem + 196608);
  half_t* QT  = (half_t*)(smem + 229376);
  half_t* G   = (half_t*)(smem + 262144);
  float*  red = (float*)(smem + 294912);

  const half_t* WembT = ws + WS_WEMBT;
  const half_t* WkT   = ws + WS_WKT;
  const half_t* WqT   = ws + WS_WQT;
  const half_t* WvT   = ws + WS_WVT;
  const half_t* WdT   = ws + WS_WDT;

  const int b    = blockIdx.x;
  const int tid  = threadIdx.x;
  const int wave = tid >> 5;
  const int lane = tid & 31;
  const int n16  = lane & 15;

  // Warm caches for the shared f16 weight streams (global_prefetch_b8).
  for (size_t off = (size_t)tid * 128; off < (size_t)WS_TOTAL * 2; off += 256 * 128)
    __builtin_prefetch((const char*)ws + off, 0, 3);

  // ---- P0: stage X[b] as f16 into LDS (vectorized) ------------------------
  const float4* Xb4 = (const float4*)(X + (size_t)b * SS * DIN);
  for (int i = tid; i < SS * DIN / 4; i += 256) {
    float4 x = Xb4[i];
    v4h h;
    h[0] = (half_t)x.x; h[1] = (half_t)x.y; h[2] = (half_t)x.z; h[3] = (half_t)x.w;
    *(v4h*)(&Xh[i * 4]) = h;
  }
  __syncthreads();

  // ---- P1: E = Xh @ WembT^T + b_emb  (256x256, Kc=64) ---------------------
  for (int t = wave; t < 256; t += 8) {
    int mt = t >> 4, nt = t & 15;
    v8f acc = {};
#pragma unroll
    for (int k0 = 0; k0 < DIN; k0 += 32) {
      v16h a = frag_a(Xh, DIN, mt * 16, k0, lane);
      v16h bm = frag_bT(WembT, DIN, nt * 16, k0, lane);
      WMMA_F16(acc, a, bm);
    }
    float bias = b_emb[nt * 16 + n16];
#pragma unroll
    for (int r = 0; r < 8; ++r) acc[r] += bias;
    store_d_f16(E, HH, mt * 16, nt * 16, acc, lane);
  }
  __syncthreads();

  // ---- P2: Sc = E @ WkT^T + b_k  (256x64, Kc=256), f32 --------------------
  for (int t = wave; t < 64; t += 8) {
    int mt = t >> 2, nt = t & 3;
    v8f acc = {};
#pragma unroll
    for (int k0 = 0; k0 < HH; k0 += 32) {
      v16h a = frag_a(E, HH, mt * 16, k0, lane);
      v16h bm = frag_bT(WkT, HH, nt * 16, k0, lane);
      WMMA_F16(acc, a, bm);
    }
    float bias = b_k[nt * 16 + n16];
#pragma unroll
    for (int r = 0; r < 8; ++r) acc[r] += bias;
    store_d_f32(Sc, KK, mt * 16, nt * 16, acc, lane);
  }
  __syncthreads();

  // ---- P3: column softmax over S; write keys^T (f16, K x S) ---------------
  if (tid < KK) {
    int k = tid;
    float m = -1e30f;
    for (int s = 0; s < SS; ++s) m = fmaxf(m, Sc[s * KK + k]);
    float sum = 0.f;
    for (int s = 0; s < SS; ++s) {
      float e = __expf(Sc[s * KK + k] - m);
      Sc[s * KK + k] = e;
      sum += e;
    }
    float inv = 1.f / sum;
    for (int s = 0; s < SS; ++s) KT[k * SS + s] = (half_t)(Sc[s * KK + k] * inv);
  }
  __syncthreads();

  // ---- P4: Q = E @ WqT^T + b_q, stored transposed (QT: K x S) -------------
  for (int t = wave; t < 64; t += 8) {
    int mt = t >> 2, nt = t & 3;
    v8f acc = {};
#pragma unroll
    for (int k0 = 0; k0 < HH; k0 += 32) {
      v16h a = frag_a(E, HH, mt * 16, k0, lane);
      v16h bm = frag_bT(WqT, HH, nt * 16, k0, lane);
      WMMA_F16(acc, a, bm);
    }
    float bias = b_q[nt * 16 + n16];
#pragma unroll
    for (int r = 0; r < 8; ++r) acc[r] += bias;
    store_d_f16_T(QT, SS, mt * 16, nt * 16, acc, lane);
  }
  __syncthreads();

  // ---- P4b/P5a: G = KT @ E in two column halves, via in-LDS transpose -----
  // ETb[h_local][s] = E[s][hsel*128 + h_local]; then E^T half is a standard
  // (N x Kc) B operand for frag_bT. Chunk rotation spreads LDS banks.
#pragma unroll
  for (int hsel = 0; hsel < 2; ++hsel) {
    // transpose half of E into ETb (each thread owns column s = tid)
#pragma unroll
    for (int j = 0; j < 16; ++j) {
      int h0 = ((j + (tid & 15)) & 15) * 8;
      v8h v = *(const v8h*)(&E[tid * HH + hsel * 128 + h0]);
#pragma unroll
      for (int e = 0; e < 8; ++e) ETb[(h0 + e) * SS + tid] = v[e];
    }
    __syncthreads();

    // G[:, hsel*128 : hsel*128+128] = KT @ E_half   (64x128, Kc=S=256)
    for (int t = wave; t < 32; t += 8) {
      int mt = t >> 3, nt = t & 7;            // 4 row tiles x 8 col tiles
      v8f acc = {};
#pragma unroll
      for (int k0 = 0; k0 < SS; k0 += 32) {
        v16h a = frag_a(KT, SS, mt * 16, k0, lane);
        v16h bm = frag_bT(ETb, SS, nt * 16, k0, lane);
        WMMA_F16(acc, a, bm);
      }
      store_d_f16(G, HH, mt * 16, hsel * 128 + nt * 16, acc, lane);
    }
    __syncthreads();
  }

  // ---- P5b: CTX = G @ WvT^T + b_v  (64x256, Kc=256) -----------------------
  for (int t = wave; t < 64; t += 8) {
    int mt = t >> 4, nt = t & 15;
    v8f acc = {};
#pragma unroll
    for (int k0 = 0; k0 < HH; k0 += 32) {
      v16h a = frag_a(G, HH, mt * 16, k0, lane);
      v16h bm = frag_bT(WvT, HH, nt * 16, k0, lane);
      WMMA_F16(acc, a, bm);
    }
    float bias = b_v[nt * 16 + n16];
#pragma unroll
    for (int r = 0; r < 8; ++r) acc[r] += bias;
    store_d_f16(CTX, HH, mt * 16, nt * 16, acc, lane);
  }
  __syncthreads();

  // ---- P6: M = QT @ WdT^T (64x256, Kc=256); out = <CTX, M> + b_d ----------
  float local = 0.f;
  const int hi = (lane >> 4) & 1;
  for (int t = wave; t < 64; t += 8) {
    int mt = t >> 4, nt = t & 15;
    v8f acc = {};
#pragma unroll
    for (int k0 = 0; k0 < SS; k0 += 32) {
      v16h a = frag_a(QT, SS, mt * 16, k0, lane);
      v16h bm = frag_bT(WdT, SS, nt * 16, k0, lane);
      WMMA_F16(acc, a, bm);
    }
#pragma unroll
    for (int r = 0; r < 8; ++r)
      local += acc[r] * (float)CTX[(mt * 16 + r + hi * 8) * HH + nt * 16 + n16];
  }
#pragma unroll
  for (int off = 16; off > 0; off >>= 1) local += __shfl_down(local, off, 32);
  if (lane == 0) red[wave] = local;
  __syncthreads();
  if (tid == 0) {
    float s = 0.f;
#pragma unroll
    for (int w = 0; w < 8; ++w) s += red[w];
    out[b] = s + b_d[0];
  }
}

// ---------------------------------------------------------------------------
extern "C" void kernel_launch(void* const* d_in, const int* in_sizes, int n_in,
                              void* d_out, int out_size, void* d_ws, size_t ws_size,
                              hipStream_t stream) {
  const float* X     = (const float*)d_in[0];
  const float* W_emb = (const float*)d_in[1];
  const float* b_emb = (const float*)d_in[2];
  const float* W_k   = (const float*)d_in[3];
  const float* b_k   = (const float*)d_in[4];
  const float* W_q   = (const float*)d_in[5];
  const float* b_q   = (const float*)d_in[6];
  const float* W_v   = (const float*)d_in[7];
  const float* b_v   = (const float*)d_in[8];
  const float* W_d   = (const float*)d_in[9];
  const float* b_d   = (const float*)d_in[10];
  float* out = (float*)d_out;
  half_t* ws = (half_t*)d_ws;

  // Weight prep (f16 transposed) into workspace.
  lambda_prep_weights<<<(WS_TOTAL + 255) / 256, 256, 0, stream>>>(
      W_emb, W_k, W_q, W_v, W_d, ws);

  // Fused per-batch kernel: 1024 blocks x 256 threads, ~288 KB dynamic LDS.
  size_t smem = 294976;
  lambda_fused_kernel<<<BB, 256, smem, stream>>>(
      X, b_emb, b_k, b_q, b_v, b_d, ws, out);
}
